// SoftFillEdgeEnhance_48747878809849
// MI455X (gfx1250) — matrix operand order repaired and checked by reading, not compiled, not run on hardware
//
#include <hip/hip_runtime.h>
#include <math.h>

#define B_  8
#define C_  128
#define CM  64
#define H_  160
#define W_  160
#define HW  (H_*W_)

typedef __attribute__((ext_vector_type(16))) _Float16 v16h;
typedef __attribute__((ext_vector_type(8)))  _Float16 v8h;
typedef __attribute__((ext_vector_type(8)))  float    v8f;

__device__ __forceinline__ float sigmoidf_(float x){ return 1.0f/(1.0f+__expf(-x)); }
__device__ __forceinline__ float siluf_(float x){ return x*sigmoidf_(x); }

__device__ __forceinline__ v16h make_frag(const _Float16* lo, const _Float16* hi){
  v8h a = *(const v8h*)lo;
  v8h b = *(const v8h*)hi;
  v16h r;
  #pragma unroll
  for (int i=0;i<8;i++){ r[i]=a[i]; r[i+8]=b[i]; }
  return r;
}

__device__ __forceinline__ v8f wmma_f16(v16h a, v16h b, v8f c){
  return __builtin_amdgcn_wmma_f32_16x16x32_f16(false, a, false, b, (short)0, c, false, false);
}

// ---------------- K0: BN folding, weight f16 conversion (+tap-major conv2 weights) ----------------
__global__ void k_prep(const float* __restrict__ fsr,
                       const float* __restrict__ w1, const float* __restrict__ w2, const float* __restrict__ w3,
                       const float* __restrict__ g1, const float* __restrict__ b1, const float* __restrict__ m1, const float* __restrict__ v1,
                       const float* __restrict__ g2, const float* __restrict__ b2, const float* __restrict__ m2, const float* __restrict__ v2,
                       const float* __restrict__ g3, const float* __restrict__ b3, const float* __restrict__ m3, const float* __restrict__ v3,
                       _Float16* __restrict__ w1h, _Float16* __restrict__ w2t, _Float16* __restrict__ w3h,
                       float* __restrict__ s1, float* __restrict__ c1,
                       float* __restrict__ s2, float* __restrict__ c2,
                       float* __restrict__ s3, float* __restrict__ c3,
                       float* __restrict__ strength){
  int tid = blockIdx.x*blockDim.x + threadIdx.x;
  int stride = gridDim.x*blockDim.x;
  for (int i=tid; i<CM*C_; i+=stride) w1h[i] = (_Float16)w1[i];
  for (int i=tid; i<CM*CM*9; i+=stride){
    int co = i/576, rem = i - co*576, ci = rem/9, t = rem - ci*9;
    w2t[t*(CM*CM) + co*CM + ci] = (_Float16)w2[i];   // [tap][co][ci]
  }
  for (int i=tid; i<C_*CM; i+=stride) w3h[i] = (_Float16)w3[i];
  if (tid < CM){
    float s = g1[tid]*rsqrtf(v1[tid]+1e-5f); s1[tid]=s; c1[tid]=b1[tid]-m1[tid]*s;
    float t = g2[tid]*rsqrtf(v2[tid]+1e-5f); s2[tid]=t; c2[tid]=b2[tid]-m2[tid]*t;
  }
  if (tid < C_){
    float s = g3[tid]*rsqrtf(v3[tid]+1e-5f); s3[tid]=s; c3[tid]=b3[tid]-m3[tid]*s;
    strength[tid] = sigmoidf_(fsr[tid]);
  }
}

// ---------------- K1: per-(b,c) plane: contrast, fill, channel weight ----------------
__global__ void __launch_bounds__(256) k_plane(const float* __restrict__ x,
                                               const float* __restrict__ strength,
                                               float* __restrict__ filled,
                                               float* __restrict__ chw){
  int plane = blockIdx.x;                 // b*C + c
  int c = plane & (C_-1);
  const float* xp = x + (size_t)plane*HW;
  float* fp = filled + (size_t)plane*HW;
  __shared__ float redA[256], redB[256];
  int tid = threadIdx.x;

  float minv = 3.4e38f, csum = 0.f;
  for (int p=tid; p<HW; p+=256){
    int h = p / W_, w = p - h*W_;
    float mx = -3.4e38f, mn = 3.4e38f;
    #pragma unroll
    for (int dh=-1; dh<=1; dh++){
      int hh = h + dh; if ((unsigned)hh >= (unsigned)H_) continue;
      #pragma unroll
      for (int dw=-1; dw<=1; dw++){
        int ww = w + dw; if ((unsigned)ww >= (unsigned)W_) continue;
        float v = xp[hh*W_+ww];
        mx = fmaxf(mx, v); mn = fminf(mn, v);
      }
    }
    minv = fminf(minv, xp[p]);
    float ct = mx - mn;
    fp[p] = ct;           // stage contrast, overwritten in phase 2
    csum += ct;
  }
  redA[tid]=minv; redB[tid]=csum; __syncthreads();
  for (int s=128; s>0; s>>=1){
    if (tid < s){ redA[tid]=fminf(redA[tid],redA[tid+s]); redB[tid]+=redB[tid+s]; }
    __syncthreads();
  }
  float baseline = redA[0];
  float thr = 0.3f*(redB[0]/(float)HW + 1e-6f);
  float st = strength[c];
  __syncthreads();

  float amax = 0.f, asum = 0.f;
  for (int p=tid; p<HW; p+=256){
    float ct = fp[p];
    float v  = xp[p];
    float m  = (ct < thr) ? 1.f : 0.f;
    float f  = v*(1.f - st*m) + baseline*st*m;
    fp[p] = f;
    float a = fabsf(f);
    amax = fmaxf(amax, a); asum += a;
  }
  redA[tid]=amax; redB[tid]=asum; __syncthreads();
  for (int s=128; s>0; s>>=1){
    if (tid < s){ redA[tid]=fmaxf(redA[tid],redA[tid+s]); redB[tid]+=redB[tid+s]; }
    __syncthreads();
  }
  if (tid==0){
    float mch = redA[0], ach = redB[0]/(float)HW;
    chw[plane] = sigmoidf_(3.0f*(mch - ach - 0.3f));
  }
}

// ---------------- K2: x_spatial = mean_c |filled| ----------------
__global__ void __launch_bounds__(256) k_spatial(const float* __restrict__ filled,
                                                 float* __restrict__ spatial){
  int idx = blockIdx.x*256 + threadIdx.x;   // b*HW + p
  int b = idx / HW, p = idx - b*HW;
  const float* base = filled + ((size_t)b*C_)*HW + p;
  float s = 0.f;
  #pragma unroll 4
  for (int c=0; c<C_; c++) s += fabsf(base[(size_t)c*HW]);
  spatial[idx] = s*(1.0f/(float)C_);
}

// ---------------- K3: edge weight + enhance (f32 in place NCHW) + f16 copy as [b][p][c] ----------------
__global__ void __launch_bounds__(256) k_enhance(const float* __restrict__ spatial,
                                                 const float* __restrict__ chw,
                                                 float* __restrict__ filled,
                                                 _Float16* __restrict__ enh16){
  int idx = blockIdx.x*256 + threadIdx.x;   // b*HW + p
  int b = idx / HW, p = idx - b*HW;
  int h = p / W_, w = p - h*W_;
  const float* sp = spatial + (size_t)b*HW;
  float mx = -3.4e38f, mn = 3.4e38f;
  #pragma unroll
  for (int dh=-1; dh<=1; dh++){
    int hh = h + dh; if ((unsigned)hh >= (unsigned)H_) continue;
    #pragma unroll
    for (int dw=-1; dw<=1; dw++){
      int ww = w + dw; if ((unsigned)ww >= (unsigned)W_) continue;
      float v = sp[hh*W_+ww];
      mx = fmaxf(mx, v); mn = fminf(mn, v);
    }
  }
  float edge = mx - mn;
  float fac = 1.0f + sigmoidf_(5.0f*(edge - 0.5f));
  float* fb = filled + ((size_t)b*C_)*HW + p;              // NCHW
  _Float16* eb = enh16 + ((size_t)(b*HW + p))*C_;          // [b][p][c]
  const float* cw = chw + b*C_;
  for (int cb=0; cb<C_; cb+=8){
    v8h hv;
    #pragma unroll
    for (int e=0; e<8; e++){
      int c = cb + e;
      float val = fb[(size_t)c*HW]*cw[c]*fac;
      fb[(size_t)c*HW] = val;
      hv[e] = (_Float16)val;
    }
    *(v8h*)(eb + cb) = hv;
  }
}

// ---------------- K4: conv1 1x1 (128->64) GEMM + BN + SiLU ----------------
// activations [b][p][128]; output o1 [b][p][64]; direct-global WMMA fragments
__global__ void __launch_bounds__(256) k_conv1(const _Float16* __restrict__ enh16,
                                               const _Float16* __restrict__ w1h,   // [64][128]
                                               const float* __restrict__ s1, const float* __restrict__ c1,
                                               _Float16* __restrict__ o1){
  int tid = threadIdx.x, blk = blockIdx.x;
  int b  = blk / (HW/64);
  int n0 = (blk - b*(HW/64))*64;
  int lane = tid & 31, wave = tid >> 5;
  int m0 = (wave & 3)*16, ng = wave >> 2;
  int kg = lane >> 4, colsel = lane & 15;
  int row = m0 + colsel;
  int col0 = n0 + ng*32 + colsel, col1 = col0 + 16;
  const _Float16* Ab = w1h + (size_t)row*C_;
  const _Float16* B0 = enh16 + ((size_t)(b*HW + col0))*C_;
  const _Float16* B1 = enh16 + ((size_t)(b*HW + col1))*C_;
  v8f acc0 = {}, acc1 = {};
  #pragma unroll
  for (int kb=0; kb<C_; kb+=32){
    v16h a  = make_frag(Ab + kb + kg*8,  Ab + kb + 16 + kg*8);
    v16h f0 = make_frag(B0 + kb + kg*8, B0 + kb + 16 + kg*8);
    v16h f1 = make_frag(B1 + kb + kg*8, B1 + kb + 16 + kg*8);
    acc0 = wmma_f16(a, f0, acc0);
    acc1 = wmma_f16(a, f1, acc1);
  }
  int ch0 = m0 + kg*8;                       // 8 consecutive out-channels
  v8h h0, h1;
  #pragma unroll
  for (int v=0; v<8; v++){
    float sc = s1[ch0+v], bi = c1[ch0+v];
    h0[v] = (_Float16)siluf_(acc0[v]*sc + bi);
    h1[v] = (_Float16)siluf_(acc1[v]*sc + bi);
  }
  *(v8h*)&o1[((size_t)(b*HW + col0))*CM + ch0] = h0;
  *(v8h*)&o1[((size_t)(b*HW + col1))*CM + ch0] = h1;
}

// ---------------- K5: conv2 3x3 (64->64) implicit GEMM + BN + SiLU ----------------
__global__ void __launch_bounds__(256) k_conv2(const _Float16* __restrict__ o1,    // [b][p][64]
                                               const _Float16* __restrict__ w2t,   // [tap][co][ci]
                                               const float* __restrict__ s2, const float* __restrict__ c2,
                                               _Float16* __restrict__ o2){         // [b][p][64]
  int tid = threadIdx.x, blk = blockIdx.x;
  int b  = blk / (HW/64);
  int n0 = (blk - b*(HW/64))*64;
  int lane = tid & 31, wave = tid >> 5;
  int m0 = (wave & 3)*16, ng = wave >> 2;
  int kg = lane >> 4, colsel = lane & 15;
  int row = m0 + colsel;
  int p0 = n0 + ng*32 + colsel, p1 = p0 + 16;
  int h0 = p0 / W_, w0 = p0 - h0*W_;
  int h1 = p1 / W_, w1 = p1 - h1*W_;
  const _Float16* obat = o1 + ((size_t)b*HW)*CM;
  v8f acc0 = {}, acc1 = {};
  v16h zf = {};
  for (int t=0; t<9; t++){
    int dh = t/3 - 1, dw = t%3 - 1;
    int hh0 = h0 + dh, ww0 = w0 + dw;
    int hh1 = h1 + dh, ww1 = w1 + dw;
    bool in0 = (unsigned)hh0 < (unsigned)H_ && (unsigned)ww0 < (unsigned)W_;
    bool in1 = (unsigned)hh1 < (unsigned)H_ && (unsigned)ww1 < (unsigned)W_;
    const _Float16* bp0 = obat + ((size_t)(hh0*W_ + ww0))*CM;
    const _Float16* bp1 = obat + ((size_t)(hh1*W_ + ww1))*CM;
    const _Float16* ap  = w2t + (size_t)t*(CM*CM) + (size_t)row*CM;
    if (t < 8) __builtin_prefetch(obat + ((size_t)(hh0*W_ + ww0 + 1))*CM, 0, 1);
    #pragma unroll
    for (int cg=0; cg<2; cg++){
      int kc0 = cg*32;
      v16h a = make_frag(ap + kc0 + kg*8, ap + kc0 + 16 + kg*8);
      v16h f0, f1;
      if (in0) f0 = make_frag(bp0 + kc0 + kg*8, bp0 + kc0 + 16 + kg*8); else f0 = zf;
      if (in1) f1 = make_frag(bp1 + kc0 + kg*8, bp1 + kc0 + 16 + kg*8); else f1 = zf;
      acc0 = wmma_f16(a, f0, acc0);
      acc1 = wmma_f16(a, f1, acc1);
    }
  }
  int ch0 = m0 + kg*8;
  v8h g0, g1;
  #pragma unroll
  for (int v=0; v<8; v++){
    float sc = s2[ch0+v], bi = c2[ch0+v];
    g0[v] = (_Float16)siluf_(acc0[v]*sc + bi);
    g1[v] = (_Float16)siluf_(acc1[v]*sc + bi);
  }
  *(v8h*)&o2[((size_t)(b*HW + p0))*CM + ch0] = g0;
  *(v8h*)&o2[((size_t)(b*HW + p1))*CM + ch0] = g1;
}

// ---------------- K6: conv3 1x1 (64->128) GEMM + BN + residual + SiLU ----------------
__global__ void __launch_bounds__(256) k_conv3(const _Float16* __restrict__ o2,    // [b][p][64]
                                               const _Float16* __restrict__ w3h,   // [128][64]
                                               const float* __restrict__ s3, const float* __restrict__ c3,
                                               const float* __restrict__ enh,      // enhanced f32 NCHW
                                               float* __restrict__ out){           // NCHW f32
  int tid = threadIdx.x, blk = blockIdx.x;
  int b  = blk / (HW/32);
  int n0 = (blk - b*(HW/32))*32;
  int lane = tid & 31, wave = tid >> 5;
  int m0 = wave*16;
  int kg = lane >> 4, colsel = lane & 15;
  int row = m0 + colsel;
  int col0 = n0 + colsel, col1 = col0 + 16;
  const _Float16* Ab = w3h + (size_t)row*CM;
  const _Float16* B0 = o2 + ((size_t)(b*HW + col0))*CM;
  const _Float16* B1 = o2 + ((size_t)(b*HW + col1))*CM;
  v8f acc0 = {}, acc1 = {};
  #pragma unroll
  for (int kb=0; kb<CM; kb+=32){
    v16h a  = make_frag(Ab + kb + kg*8,  Ab + kb + 16 + kg*8);
    v16h f0 = make_frag(B0 + kb + kg*8, B0 + kb + 16 + kg*8);
    v16h f1 = make_frag(B1 + kb + kg*8, B1 + kb + 16 + kg*8);
    acc0 = wmma_f16(a, f0, acc0);
    acc1 = wmma_f16(a, f1, acc1);
  }
  #pragma unroll
  for (int v=0; v<8; v++){
    int r = m0 + v + kg*8;
    float sc = s3[r], bi = c3[r];
    size_t i0 = ((size_t)(b*C_ + r))*HW + col0;
    size_t i1 = ((size_t)(b*C_ + r))*HW + col1;
    out[i0] = siluf_(acc0[v]*sc + bi + enh[i0]);
    out[i1] = siluf_(acc1[v]*sc + bi + enh[i1]);
  }
}

// ---------------- launch ----------------
extern "C" void kernel_launch(void* const* d_in, const int* in_sizes, int n_in,
                              void* d_out, int out_size, void* d_ws, size_t ws_size,
                              hipStream_t stream) {
  const float* x   = (const float*)d_in[0];
  const float* fsr = (const float*)d_in[1];
  const float* w1  = (const float*)d_in[2];
  const float* g1  = (const float*)d_in[3];
  const float* b1  = (const float*)d_in[4];
  const float* m1  = (const float*)d_in[5];
  const float* v1  = (const float*)d_in[6];
  const float* w2  = (const float*)d_in[7];
  const float* g2  = (const float*)d_in[8];
  const float* b2  = (const float*)d_in[9];
  const float* m2  = (const float*)d_in[10];
  const float* v2  = (const float*)d_in[11];
  const float* w3  = (const float*)d_in[12];
  const float* g3  = (const float*)d_in[13];
  const float* b3  = (const float*)d_in[14];
  const float* m3  = (const float*)d_in[15];
  const float* v3  = (const float*)d_in[16];
  float* out = (float*)d_out;

  char* ws = (char*)d_ws;
  size_t off = 0;
  auto alloc = [&](size_t bytes) -> char* {
    char* p = ws + off;
    off = (off + bytes + 255) & ~(size_t)255;
    return p;
  };
  float*    filled  = (float*)   alloc((size_t)B_*C_*HW*sizeof(float));     // -> enhanced f32 (NCHW) in place
  _Float16* enh16   = (_Float16*)alloc((size_t)B_*C_*HW*sizeof(_Float16));  // [b][p][c]
  float*    spatial = (float*)   alloc((size_t)B_*HW*sizeof(float));
  _Float16* o1      = (_Float16*)alloc((size_t)B_*CM*HW*sizeof(_Float16));  // [b][p][c]
  _Float16* o2      = (_Float16*)alloc((size_t)B_*CM*HW*sizeof(_Float16));  // [b][p][c]
  _Float16* w1h     = (_Float16*)alloc((size_t)CM*C_*sizeof(_Float16));
  _Float16* w2t     = (_Float16*)alloc((size_t)CM*CM*9*sizeof(_Float16));
  _Float16* w3h     = (_Float16*)alloc((size_t)C_*CM*sizeof(_Float16));
  float*    s1      = (float*)   alloc(CM*sizeof(float));
  float*    c1      = (float*)   alloc(CM*sizeof(float));
  float*    s2      = (float*)   alloc(CM*sizeof(float));
  float*    c2      = (float*)   alloc(CM*sizeof(float));
  float*    s3      = (float*)   alloc(C_*sizeof(float));
  float*    c3      = (float*)   alloc(C_*sizeof(float));
  float*    chw     = (float*)   alloc((size_t)B_*C_*sizeof(float));
  float*    strength= (float*)   alloc(C_*sizeof(float));

  k_prep<<<144, 256, 0, stream>>>(fsr, w1, w2, w3,
                                  g1, b1, m1, v1, g2, b2, m2, v2, g3, b3, m3, v3,
                                  w1h, w2t, w3h, s1, c1, s2, c2, s3, c3, strength);
  k_plane<<<B_*C_, 256, 0, stream>>>(x, strength, filled, chw);
  k_spatial<<<(B_*HW)/256, 256, 0, stream>>>(filled, spatial);
  k_enhance<<<(B_*HW)/256, 256, 0, stream>>>(spatial, chw, filled, enh16);
  k_conv1<<<B_*(HW/64), 256, 0, stream>>>(enh16, w1h, s1, c1, o1);
  k_conv2<<<B_*(HW/64), 256, 0, stream>>>(o1, w2t, s2, c2, o2);
  k_conv3<<<B_*(HW/32), 256, 0, stream>>>(o2, w3h, s3, c3, filled, out);
}
